// KernelClassifier_42958262894772
// MI455X (gfx1250) — compile-verified, matrix operand order checked by reading, not run on hardware
//
#include <hip/hip_runtime.h>
#include <hip/hip_bf16.h>
#include <stddef.h>

// ---------------- problem constants ----------------
constexpr int Bq    = 1024;     // query batch
constexpr int Nn    = 50000;    // train set
constexpr int DIN   = 768;
constexpr int DP    = 128;      // projection dim
constexpr int Cc    = 100;      // classes
constexpr int NPAD  = 50176;    // 392 * 128
constexpr int NTILES = NPAD / 128;   // 392
constexpr float INV_2BW2 = 1.0f / 256.0f;   // 2*BW2 = 2*128
constexpr float EPSV = 1e-10f;

typedef __attribute__((ext_vector_type(16))) __bf16 v16bf;
typedef __attribute__((ext_vector_type(8)))  __bf16 v8bf;
typedef __attribute__((ext_vector_type(8)))  float  v8f;
typedef __attribute__((ext_vector_type(4)))  float  v4f;

#define WMMA_BF16(a, b, c) \
    __builtin_amdgcn_wmma_f32_16x16x32_bf16(false, (a), false, (b), (short)0, (c), false, false)

static __device__ inline v16bf cat8(v8bf lo, v8bf hi) {
    return __builtin_shufflevector(lo, hi, 0,1,2,3,4,5,6,7,8,9,10,11,12,13,14,15);
}

// ---------------- gfx1250 async-to-LDS path (guarded) ----------------
#if defined(__has_builtin)
#  if __has_builtin(__builtin_amdgcn_global_load_async_to_lds_b128)
#    define HAVE_ASYNC 1
#  endif
#endif
#ifndef HAVE_ASYNC
#  define HAVE_ASYNC 0
#endif

#if HAVE_ASYNC
// builtin signature (from hipcc diagnostic): (int4 addrspace(1)*, int4 addrspace(3)*, imm, imm)
typedef int v4i __attribute__((vector_size(16)));
typedef __attribute__((address_space(1))) v4i* gas_v4i;
typedef __attribute__((address_space(3))) v4i* las_v4i;
static __device__ inline void async_cp16(const void* g, void* l) {
    // per-lane 16B copy global -> LDS, tracked by ASYNCcnt (no VGPR staging)
    __builtin_amdgcn_global_load_async_to_lds_b128((gas_v4i)g, (las_v4i)l, 0, 0);
}
#  if __has_builtin(__builtin_amdgcn_s_wait_asynccnt)
#    define WAIT_ASYNC(n) __builtin_amdgcn_s_wait_asynccnt((unsigned short)(n))
#  else
#    define WAIT_ASYNC(n) asm volatile("s_wait_asynccnt %0" :: "n"(n) : "memory")
#  endif
#endif

// ---------------- kernel 1: init workspace ----------------
__global__ void k_init(float* __restrict__ pred, int* __restrict__ cnt,
                       __bf16* __restrict__ pXh, float* __restrict__ nX,
                       int* __restrict__ lab) {
    int i = blockIdx.x * blockDim.x + threadIdx.x;
    int stride = gridDim.x * blockDim.x;
    for (int k = i; k < Bq * Cc; k += stride) pred[k] = 0.0f;
    for (int k = i; k < Cc; k += stride) cnt[k] = 0;
    const int padN = NPAD - Nn;                      // 176
    for (int k = i; k < padN * DP; k += stride) pXh[(size_t)Nn * DP + k] = (__bf16)0.0f;
    for (int k = i; k < padN; k += stride) { nX[Nn + k] = 1e30f; lab[Nn + k] = 0; }
}

// ---------------- kernel 2: label ranks + class histogram ----------------
__global__ void k_labels(const float* __restrict__ SorP, const int* __restrict__ Y,
                         int* __restrict__ lab, int* __restrict__ cnt) {
    int row  = blockIdx.x * 8 + (threadIdx.x >> 5);
    int lane = threadIdx.x & 31;
    if (row >= Nn) return;
    int yi = Y[row];
    const float* s = SorP + (size_t)row * Cc;
    float v = s[yi];
    int r = 0;
    for (int j = lane; j < Cc; j += 32) {
        float sj = s[j];
        r += (sj > v) || (sj == v && j < yi);
    }
    for (int m = 16; m; m >>= 1) r += __shfl_down(r, m, 32);
    if (lane == 0) { lab[row] = r; atomicAdd(&cnt[r], 1); }
}

// ---------------- kernel 3: projection GEMM (fp32 -> bf16) ----------------
__global__ void k_project(const float* __restrict__ src, int M,
                          const float* __restrict__ Wp, const float* __restrict__ bp,
                          __bf16* __restrict__ dstH, float* __restrict__ norms) {
    __shared__ __align__(32) __bf16 ldsB[8][32][16];   // Wp k-slab in B-fragment order

    const int tid  = threadIdx.x;
    const int wave = tid >> 5;
    const int lane = tid & 31;
    const int lan  = lane & 15;
    const int hsel = lane >> 4;
    const int Mbase = blockIdx.x * 128;
    const int mrow  = Mbase + wave * 16 + lan;
    const int mA    = mrow < M ? mrow : (M - 1);

    v8f acc[8];
    #pragma unroll
    for (int nt = 0; nt < 8; ++nt) acc[nt] = (v8f)0.0f;

    for (int k0 = 0; k0 < DIN; k0 += 32) {
        __syncthreads();
        {
            int nt = tid >> 5, l = tid & 31;
            int ls = l & 15, hs = l >> 4;
            v16bf tmp;
            #pragma unroll
            for (int s = 0; s < 16; ++s) {
                int k = (s < 8 ? s : s + 8) + hs * 8;
                tmp[s] = (__bf16)Wp[(size_t)(k0 + k) * DP + nt * 16 + ls];
            }
            *(v16bf*)&ldsB[nt][l][0] = tmp;
        }
        __syncthreads();

        const float* base = src + (size_t)mA * DIN + k0 + hsel * 8;
        v4f f0 = *(const v4f*)(base +  0);
        v4f f1 = *(const v4f*)(base +  4);
        v4f f2 = *(const v4f*)(base + 16);
        v4f f3 = *(const v4f*)(base + 20);
        v16bf av;
        #pragma unroll
        for (int i = 0; i < 4; ++i) {
            av[i]      = (__bf16)f0[i];
            av[4 + i]  = (__bf16)f1[i];
            av[8 + i]  = (__bf16)f2[i];
            av[12 + i] = (__bf16)f3[i];
        }
        #pragma unroll
        for (int nt = 0; nt < 8; ++nt) {
            v16bf bv = *(const v16bf*)&ldsB[nt][lane][0];
            acc[nt] = WMMA_BF16(av, bv, acc[nt]);
        }
    }

    float bpv[8];
    #pragma unroll
    for (int nt = 0; nt < 8; ++nt) bpv[nt] = bp[nt * 16 + lan];
    #pragma unroll
    for (int nt = 0; nt < 8; ++nt)
        #pragma unroll
        for (int r = 0; r < 8; ++r) acc[nt][r] += bpv[nt];

    #pragma unroll
    for (int r = 0; r < 8; ++r) {
        float p = 0.0f;
        #pragma unroll
        for (int nt = 0; nt < 8; ++nt) { float v = acc[nt][r]; p += v * v; }
        for (int m = 1; m < 16; m <<= 1) p += __shfl_xor(p, m, 32);
        int row = Mbase + wave * 16 + hsel * 8 + r;
        if (lan == 0 && row < M) norms[row] = p;
    }

    #pragma unroll
    for (int nt = 0; nt < 8; ++nt)
        #pragma unroll
        for (int r = 0; r < 8; ++r) {
            int row = Mbase + wave * 16 + hsel * 8 + r;
            if (row < M) dstH[(size_t)row * DP + nt * 16 + lan] = (__bf16)acc[nt][r];
        }
}

// ---------------- kernel 4: fused distance -> exp -> label scatter ----------------
// B tiles (128 x 128 bf16 = 32KB) staged into LDS via async-copy, double buffered:
// all 8 waves of the block share one copy (8x less L2 traffic than direct loads).
constexpr int CP = 104;                               // padded class stride
constexpr int TILEB      = 128 * DP * 2;              // 32768 B per B-tile
constexpr int SMEM_PRED  = 128 * CP * 4;              // 53248 B
constexpr int SMEM_TOTAL = SMEM_PRED + 2 * TILEB;     // 118784 B (dynamic LDS, <=320KB/WG)

__global__ void k_distance(const __bf16* __restrict__ pxh, const __bf16* __restrict__ pXh,
                           const float* __restrict__ nx, const float* __restrict__ nXv,
                           const int* __restrict__ lab, float* __restrict__ pred) {
    extern __shared__ __align__(16) char smem[];
    float (*predS)[CP] = (float (*)[CP])smem;
    char* bufB = smem + SMEM_PRED;

    const int tid  = threadIdx.x;
    const int wave = tid >> 5;
    const int lane = tid & 31;
    const int lan  = lane & 15;
    const int hsel = lane >> 4;
    const int bbase = blockIdx.y * 128;
    const int STR   = (int)gridDim.x;

    for (int i = tid; i < 128 * CP; i += 256) ((float*)predS)[i] = 0.0f;

    // hoisted A fragments (K = 128 -> 4 steps of 32) + query norms
    const int mrow = bbase + wave * 16 + lan;
    v16bf a[4];
    #pragma unroll
    for (int kk = 0; kk < 4; ++kk) {
        const __bf16* p = pxh + (size_t)mrow * DP + kk * 32 + hsel * 8;
        a[kk] = cat8(*(const v8bf*)p, *(const v8bf*)(p + 16));
    }
    float nxr[8];
    #pragma unroll
    for (int r = 0; r < 8; ++r) nxr[r] = nx[bbase + wave * 16 + hsel * 8 + r];

#if HAVE_ASYNC
    {   // prologue: async-stage first tile into buffer 0
        const char* g = (const char*)(pXh + (size_t)blockIdx.x * 128 * DP);
        #pragma unroll
        for (int j = 0; j < 8; ++j)
            async_cp16(g + tid * 16 + j * 4096, bufB + tid * 16 + j * 4096);
    }
#endif

    int p = 0;
    for (int ntile = blockIdx.x; ntile < NTILES; ntile += STR) {
#if HAVE_ASYNC
        const int nxt = ntile + STR;
        if (nxt < NTILES) {   // overlap next tile's copy with this tile's WMMAs
            const char* g = (const char*)(pXh + (size_t)nxt * 128 * DP);
            char* l = bufB + (p ^ 1) * TILEB;
            #pragma unroll
            for (int j = 0; j < 8; ++j)
                async_cp16(g + tid * 16 + j * 4096, l + tid * 16 + j * 4096);
            WAIT_ASYNC(8);    // in-order: 8 oldest (current tile) have landed
        } else {
            WAIT_ASYNC(0);
        }
        __syncthreads();      // all waves' copies visible
        const __bf16* bt = (const __bf16*)(bufB + p * TILEB);
#else
        __syncthreads();      // prev readers of bufB done
        {   // synchronous fallback: reg-staged copy of the tile
            const char* g = (const char*)(pXh + (size_t)ntile * 128 * DP);
            #pragma unroll
            for (int j = 0; j < 8; ++j) {
                v4f tmp = *(const v4f*)(g + tid * 16 + j * 4096);
                *(v4f*)(bufB + tid * 16 + j * 4096) = tmp;
            }
        }
        __syncthreads();
        const __bf16* bt = (const __bf16*)bufB;
#endif
        const int n0 = ntile * 128;

        int   labc[8];
        float nXc[8];
        #pragma unroll
        for (int nt = 0; nt < 8; ++nt) {
            int n = n0 + nt * 16 + lan;
            labc[nt] = lab[n];
            nXc[nt]  = nXv[n];
        }

        v8f acc[8];
        #pragma unroll
        for (int nt = 0; nt < 8; ++nt) acc[nt] = (v8f)0.0f;

        #pragma unroll
        for (int kk = 0; kk < 4; ++kk)
            #pragma unroll
            for (int nt = 0; nt < 8; ++nt) {
                // B = pX^T: tile row (nt*16+lan) holds column n0+nt*16+lan
                const __bf16* pB = bt + (nt * 16 + lan) * DP + kk * 32 + hsel * 8;
                v16bf bv = cat8(*(const v8bf*)pB, *(const v8bf*)(pB + 16));
                acc[nt] = WMMA_BF16(a[kk], bv, acc[nt]);
            }

        #pragma unroll
        for (int nt = 0; nt < 8; ++nt)
            #pragma unroll
            for (int r = 0; r < 8; ++r) {
                float s   = acc[nt][r];
                float sqd = nxr[r] + nXc[nt] - 2.0f * s;
                float kis = __expf(-sqd * INV_2BW2);
                kis = (kis < 1.0f - 1e-10f) ? kis : 0.0f;   // drop_self
                int rowl = wave * 16 + hsel * 8 + r;
                atomicAdd(&predS[rowl][labc[nt]], kis);
            }

        __syncthreads();      // all reads of buf[p] done before it is re-filled
        p ^= 1;
    }

    for (int i = tid; i < 128 * Cc; i += 256) {
        int row = i / Cc, c = i - row * Cc;
        float v = predS[row][c];
        if (v != 0.0f) atomicAdd(&pred[(size_t)(bbase + row) * Cc + c], v);
    }
}

// ---------------- kernel 5: normalize + per-row permutation gather ----------------
__global__ void k_finalize(const float* __restrict__ pred, const int* __restrict__ cnt,
                           const float* __restrict__ SorPq, float* __restrict__ out) {
    const int b = blockIdx.x, t = threadIdx.x;
    __shared__ float pr[112];
    __shared__ float sq[112];
    __shared__ float rsum;
    if (t < Cc) {
        pr[t] = pred[(size_t)b * Cc + t];
        sq[t] = SorPq[(size_t)b * Cc + t];
    }
    __syncthreads();
    if (t == 0) {
        float s = 0.0f;
        for (int j = 0; j < Cc; ++j) s += pr[j];
        rsum = s;
    }
    __syncthreads();
    if (t < Cc) {
        float v = sq[t];
        int rank = 0;
        for (int j = 0; j < Cc; ++j) {
            float sj = sq[j];
            rank += (sj > v) || (sj == v && j < t);
        }
        float val;
        if (rsum > EPSV) val = pr[rank] / rsum;
        else             val = (float)cnt[rank] / (float)Nn;
        out[(size_t)b * Cc + t] = val;
    }
}

// ---------------- host entry ----------------
extern "C" void kernel_launch(void* const* d_in, const int* in_sizes, int n_in,
                              void* d_out, int out_size, void* d_ws, size_t ws_size,
                              hipStream_t stream) {
    const float* x     = (const float*)d_in[0];
    const float* X     = (const float*)d_in[1];
    const float* Wp    = (const float*)d_in[2];
    const float* bp    = (const float*)d_in[3];
    const int*   Y     = (const int*)  d_in[4];
    const float* SorPt = (const float*)d_in[5];
    const float* SorPq = (const float*)d_in[6];
    float* out = (float*)d_out;

    char* ws = (char*)d_ws;
    size_t off = 0;
    float*  pred = (float*)(ws + off);  off += (size_t)Bq * Cc * 4;
    int*    cnt  = (int*)  (ws + off);  off += 512;
    float*  nx   = (float*)(ws + off);  off += (size_t)Bq * 4;
    float*  nX   = (float*)(ws + off);  off += (size_t)NPAD * 4;
    int*    lab  = (int*)  (ws + off);  off += (size_t)NPAD * 4;
    __bf16* pxh  = (__bf16*)(ws + off); off += (size_t)Bq * DP * 2;
    __bf16* pXh  = (__bf16*)(ws + off); off += (size_t)NPAD * DP * 2;
    (void)ws_size; (void)in_sizes; (void)n_in; (void)out_size;

    k_init<<<256, 256, 0, stream>>>(pred, cnt, pXh, nX, lab);
    k_labels<<<(Nn + 7) / 8, 256, 0, stream>>>(SorPt, Y, lab, cnt);
    k_project<<<Bq / 128, 256, 0, stream>>>(x, Bq, Wp, bp, pxh, nx);
    k_project<<<(Nn + 127) / 128, 256, 0, stream>>>(X, Nn, Wp, bp, pXh, nX);
    k_distance<<<dim3(32, Bq / 128), 256, SMEM_TOTAL, stream>>>(pxh, pXh, nx, nX, lab, pred);
    k_finalize<<<Bq, 128, 0, stream>>>(pred, cnt, SorPq, out);
}
